// MoME88_21191368639292
// MI455X (gfx1250) — matrix operand order repaired and checked by reading, not compiled, not run on hardware
//
#include <hip/hip_runtime.h>

#define B_    4
#define T_    2048
#define DIM_  1024
#define H_    64
#define NS_   32
#define K_    8
#define HV_   32
#define KEYD_ 2048
#define VALD_ 2048
#define BT_   (B_ * T_)

#define BM_ 128
#define BN_ 128
#define BK_ 32

typedef __attribute__((ext_vector_type(16))) __bf16 v16bf;
typedef __attribute__((ext_vector_type(8)))  float  v8f;

struct U4 { unsigned x, y, z, w; };
union FragBF { v16bf v; U4 u[2]; };

// ---------------------------------------------------------------------------
// f32 -> bf16 (round to nearest even)
// ---------------------------------------------------------------------------
__global__ void f32_to_bf16_kernel(const float* __restrict__ in,
                                   unsigned short* __restrict__ out, int n) {
    int i = blockIdx.x * blockDim.x + threadIdx.x;
    if (i < n) {
        unsigned u = __float_as_uint(in[i]);
        out[i] = (unsigned short)((u + 0x7FFFu + ((u >> 16) & 1u)) >> 16);
    }
}

// ---------------------------------------------------------------------------
// bf16 WMMA GEMM: C[M,N] = A[M,Kd] * B[N,Kd]^T, f32 accumulate.
// 8 waves per block, 128x128 block tile, K-chunk 32, double-buffered LDS
// filled with global_load_async_to_lds_b128 (ASYNCcnt-pipelined).
// Per wave: 32x64 micro-tile = 2x4 v_wmma_f32_16x16x32_bf16 frags.
// ---------------------------------------------------------------------------
__global__ __launch_bounds__(256)
void gemm_bf16_wmma(const unsigned short* __restrict__ A,
                    const unsigned short* __restrict__ Bm,
                    float* __restrict__ C, int M, int N, int Kd) {
    __shared__ unsigned short ldsA[2][BM_ * BK_];
    __shared__ unsigned short ldsB[2][BN_ * BK_];

    const int tid  = threadIdx.x;
    const int lane = tid & 31;
    const int wave = tid >> 5;      // 0..7
    const int wm   = wave >> 1;     // 0..3  (M sub-tile)
    const int wn   = wave & 1;      // 0..1  (N sub-tile)
    const int g    = lane >> 4;
    const int r    = lane & 15;

    const int nbn = N / BN_;
    const int bm0 = (blockIdx.x / nbn) * BM_;
    const int bn0 = (blockIdx.x % nbn) * BN_;

    const unsigned ldsA_base = (unsigned)(uintptr_t)&ldsA[0][0];
    const unsigned ldsB_base = (unsigned)(uintptr_t)&ldsB[0][0];

    // Stage one K-chunk (128x32 of A, 128x32 of B) into LDS buffer `buf`.
    // 512 16B transfers per matrix; each thread issues 2 for A + 2 for B
    // => 4 async instructions per wave per chunk.
    auto stage = [&](int buf, int k0) {
#pragma unroll
        for (int rr = 0; rr < 2; rr++) {
            int s   = tid + rr * 256;   // 0..511
            int row = s >> 2;           // 0..127
            int ce  = (s & 3) * 8;      // bf16 element offset within chunk
            const unsigned short* ga = A + (size_t)(bm0 + row) * Kd + k0 + ce;
            unsigned la = ldsA_base + (unsigned)(buf * BM_ * BK_ + row * BK_ + ce) * 2u;
            asm volatile("global_load_async_to_lds_b128 %0, %1, off"
                         :: "v"(la), "v"(ga) : "memory");
            const unsigned short* gb = Bm + (size_t)(bn0 + row) * Kd + k0 + ce;
            unsigned lb = ldsB_base + (unsigned)(buf * BN_ * BK_ + row * BK_ + ce) * 2u;
            asm volatile("global_load_async_to_lds_b128 %0, %1, off"
                         :: "v"(lb), "v"(gb) : "memory");
        }
    };

    v8f acc[2][4];
#pragma unroll
    for (int m = 0; m < 2; m++)
#pragma unroll
        for (int n = 0; n < 4; n++)
#pragma unroll
            for (int i = 0; i < 8; i++) acc[m][n][i] = 0.f;

    const int nK = Kd / BK_;
    int buf = 0;
    stage(0, 0);
    for (int kc = 0; kc < nK; kc++) {
        if (kc + 1 < nK) {
            stage(buf ^ 1, (kc + 1) * BK_);
            asm volatile("s_wait_asynccnt 0x4" ::: "memory"); // current chunk landed
        } else {
            asm volatile("s_wait_asynccnt 0x0" ::: "memory");
        }
        __syncthreads();   // every wave's async data visible to all

        FragBF a[2], b[4];
        // A frag: lane half g=0 -> K {0..7,16..23}, g=1 -> {8..15,24..31}
#pragma unroll
        for (int m = 0; m < 2; m++) {
            const unsigned short* ap =
                &ldsA[buf][(wm * 32 + m * 16 + r) * BK_] + g * 8;
            a[m].u[0] = *(const U4*)(ap);
            a[m].u[1] = *(const U4*)(ap + 16);
        }
        // B frag: lane half g=0 -> K {0..7,16..23}, g=1 -> {8..15,24..31}
#pragma unroll
        for (int n = 0; n < 4; n++) {
            const unsigned short* bp =
                &ldsB[buf][(wn * 64 + n * 16 + r) * BK_] + g * 8;
            b[n].u[0] = *(const U4*)(bp);
            b[n].u[1] = *(const U4*)(bp + 16);
        }
#pragma unroll
        for (int m = 0; m < 2; m++)
#pragma unroll
            for (int n = 0; n < 4; n++)
                acc[m][n] = __builtin_amdgcn_wmma_f32_16x16x32_bf16(
                    false, a[m].v, false, b[n].v, (short)0, acc[m][n],
                    false, false);
        __syncthreads();   // all reads done before next overwrite of buf^1
        buf ^= 1;
    }

    // C/D layout: VGPR rr -> row rr (lanes 0-15) / rr+8 (lanes 16-31)
#pragma unroll
    for (int m = 0; m < 2; m++)
#pragma unroll
        for (int n = 0; n < 4; n++)
#pragma unroll
            for (int rr = 0; rr < 8; rr++) {
                int row = bm0 + wm * 32 + m * 16 + g * 8 + rr;
                int col = bn0 + wn * 64 + n * 16 + r;
                C[(size_t)row * N + col] = acc[m][n][rr];
            }
}

// ---------------------------------------------------------------------------
// Router: softmax over 64 logits, top-8 + renorm, mamba2 decay. One wave/token.
// ra row layout: cols 0..63 = x.Wr logits, 64..127 = x.Wa
// ---------------------------------------------------------------------------
__global__ __launch_bounds__(32)
void router_kernel(const float* __restrict__ ra, const float* __restrict__ A_log,
                   const float* __restrict__ dt_bias, int* __restrict__ topi,
                   float* __restrict__ rw, float* __restrict__ decay) {
    int t = blockIdx.x;
    int lane = threadIdx.x;
    const float* row = ra + (size_t)t * 128;
    float l0 = row[lane], l1 = row[lane + 32];

    float m = fmaxf(l0, l1);
    for (int o = 16; o > 0; o >>= 1) m = fmaxf(m, __shfl_xor(m, o, 32));
    float e0 = expf(l0 - m), e1 = expf(l1 - m);
    float s = e0 + e1;
    for (int o = 16; o > 0; o >>= 1) s += __shfl_xor(s, o, 32);
    float pv[2] = {e0 / s, e1 / s};

    float total = 0.f, sel_p = 0.f;
    int sel_i = 0;
    for (int j = 0; j < 8; j++) {
        float lv; int li;
        if (pv[0] >= pv[1]) { lv = pv[0]; li = lane; }
        else                { lv = pv[1]; li = lane + 32; }
        for (int o = 16; o > 0; o >>= 1) {
            float ov = __shfl_xor(lv, o, 32);
            int   oi = __shfl_xor(li, o, 32);
            if (ov > lv || (ov == lv && oi < li)) { lv = ov; li = oi; }
        }
        total += lv;
        if (lane == j) { sel_p = lv; sel_i = li; }
        if (li == lane) pv[0] = -1.f;
        else if (li == lane + 32) pv[1] = -1.f;
    }
    if (lane < 8) {
        topi[(size_t)t * K_ + lane] = sel_i;
        rw[(size_t)t * K_ + lane] = sel_p / total;
    }
#pragma unroll
    for (int hh = 0; hh < 2; hh++) {
        int h = lane + hh * 32;
        float z = row[64 + h] + dt_bias[h];
        float sp = (z > 20.f) ? z : log1pf(expf(z));
        decay[(size_t)t * H_ + h] = expf(-expf(A_log[h]) * sp);
    }
}

// ---------------------------------------------------------------------------
// silu + l2norm over contiguous groups of 32 (one wave per group)
// ---------------------------------------------------------------------------
__global__ void silu_l2norm_kernel(float* __restrict__ buf, int ngroups) {
    int gid = blockIdx.x * (blockDim.x >> 5) + (threadIdx.x >> 5);
    int lane = threadIdx.x & 31;
    if (gid >= ngroups) return;
    float x = buf[(size_t)gid * 32 + lane];
    float sg = x / (1.f + expf(-x));
    float ss = sg * sg;
    for (int o = 16; o > 0; o >>= 1) ss += __shfl_xor(ss, o, 32);
    buf[(size_t)gid * 32 + lane] = sg * rsqrtf(ss + 1e-6f);
}

__global__ void silu_kernel(float* __restrict__ buf, int n) {
    int i = blockIdx.x * blockDim.x + threadIdx.x;
    if (i < n) { float x = buf[i]; buf[i] = x / (1.f + expf(-x)); }
}

// ---------------------------------------------------------------------------
// Recurrent scan: one wave per (b, slot) chain; lane = HV column; 32 state
// rows in VGPRs; v_readlane broadcasts; next timestep's dependent gather
// (idx -> k/q/v/d/w) prefetched one iteration ahead; y split into 4 chains.
// ---------------------------------------------------------------------------
__global__ __launch_bounds__(32)
void scan_kernel(const float* __restrict__ qb, const float* __restrict__ kb,
                 const float* __restrict__ vb, const float* __restrict__ decay,
                 const int* __restrict__ topi, const float* __restrict__ rw,
                 const float* __restrict__ S0, float* __restrict__ yslot,
                 float* __restrict__ Sout) {
    int chain = blockIdx.x;      // 0..B_*K_-1
    int b = chain / K_;
    int j = chain % K_;
    int lane = threadIdx.x;      // v index
    float S[NS_];
#pragma unroll
    for (int n = 0; n < NS_; n++)
        S[n] = S0[(((size_t)b * K_ + j) * NS_ + n) * HV_ + lane];

    size_t tok0 = (size_t)b * T_;
    int   idx = topi[tok0 * K_ + j];
    float w   = rw[tok0 * K_ + j];
    float d   = decay[tok0 * H_ + idx];
    float kl  = kb[tok0 * KEYD_ + idx * NS_ + lane];
    float ql  = qb[tok0 * KEYD_ + idx * NS_ + lane];
    float vl  = vb[tok0 * VALD_ + idx * HV_ + lane];

    for (int t = 0; t < T_; t++) {
        size_t tok = tok0 + t;
        // prefetch timestep t+1 (overlaps the dependent-load chain with VALU)
        int idx2 = 0; float w2 = 0.f, d2 = 0.f, kl2 = 0.f, ql2 = 0.f, vl2 = 0.f;
        if (t + 1 < T_) {
            size_t tk = tok + 1;
            idx2 = topi[tk * K_ + j];
            w2   = rw[tk * K_ + j];
            d2   = decay[tk * H_ + idx2];
            kl2  = kb[tk * KEYD_ + idx2 * NS_ + lane];
            ql2  = qb[tk * KEYD_ + idx2 * NS_ + lane];
            vl2  = vb[tk * VALD_ + idx2 * HV_ + lane];
        }
        float y0 = 0.f, y1 = 0.f, y2 = 0.f, y3 = 0.f;
#pragma unroll
        for (int n = 0; n < NS_; n += 4) {
            float ka = __shfl(kl, n + 0, 32), qa = __shfl(ql, n + 0, 32);
            float kbx = __shfl(kl, n + 1, 32), qbx = __shfl(ql, n + 1, 32);
            float kc = __shfl(kl, n + 2, 32), qc = __shfl(ql, n + 2, 32);
            float kd = __shfl(kl, n + 3, 32), qd = __shfl(ql, n + 3, 32);
            S[n + 0] = fmaf(ka, vl, d * S[n + 0]); y0 = fmaf(qa, S[n + 0], y0);
            S[n + 1] = fmaf(kbx, vl, d * S[n + 1]); y1 = fmaf(qbx, S[n + 1], y1);
            S[n + 2] = fmaf(kc, vl, d * S[n + 2]); y2 = fmaf(qc, S[n + 2], y2);
            S[n + 3] = fmaf(kd, vl, d * S[n + 3]); y3 = fmaf(qd, S[n + 3], y3);
        }
        yslot[(tok * K_ + j) * HV_ + lane] = w * ((y0 + y1) + (y2 + y3));
        idx = idx2; w = w2; d = d2; kl = kl2; ql = ql2; vl = vl2;
    }
#pragma unroll
    for (int n = 0; n < NS_; n++)
        Sout[(((size_t)b * K_ + j) * NS_ + n) * HV_ + lane] = S[n];
}

// ---------------------------------------------------------------------------
// Final: y[t,d] = sum_v (sum_j yslot[t,j,v]) * Wo[d,v]
// ---------------------------------------------------------------------------
__global__ __launch_bounds__(256)
void out_proj_kernel(const float* __restrict__ yslot,
                     const float* __restrict__ Wo, float* __restrict__ y) {
    int t = blockIdx.x;
    int tid = threadIdx.x;
    __shared__ float osum[HV_];
    if (tid < HV_) {
        float s = 0.f;
#pragma unroll
        for (int j = 0; j < K_; j++)
            s += yslot[((size_t)t * K_ + j) * HV_ + tid];
        osum[tid] = s;
    }
    __syncthreads();
    for (int d = tid; d < DIM_; d += 256) {
        const float* wr = Wo + (size_t)d * HV_;
        float acc = 0.f;
#pragma unroll
        for (int v = 0; v < HV_; v++) acc = fmaf(osum[v], wr[v], acc);
        y[(size_t)t * DIM_ + d] = acc;
    }
}

// ---------------------------------------------------------------------------
extern "C" void kernel_launch(void* const* d_in, const int* in_sizes, int n_in,
                              void* d_out, int out_size, void* d_ws,
                              size_t ws_size, hipStream_t stream) {
    (void)in_sizes; (void)n_in; (void)out_size; (void)ws_size;
    const float* x       = (const float*)d_in[0];
    const float* Wr      = (const float*)d_in[1];
    const float* Wq      = (const float*)d_in[2];
    const float* Wk      = (const float*)d_in[3];
    const float* Wv      = (const float*)d_in[4];
    const float* Wa      = (const float*)d_in[5];
    const float* A_log   = (const float*)d_in[6];
    const float* dt_bias = (const float*)d_in[7];
    const float* Wo      = (const float*)d_in[8];
    const float* S0      = (const float*)d_in[9];

    float* out_y = (float*)d_out;                  // [BT, DIM]
    float* out_S = out_y + (size_t)BT_ * DIM_;     // [B, K, NS, HV]

    char* ws = (char*)d_ws;
    size_t off = 0;
    auto take = [&](size_t bytes) -> char* {
        char* p = ws + off;
        off += (bytes + 255) & ~(size_t)255;
        return p;
    };
    unsigned short* xb   = (unsigned short*)take((size_t)BT_ * DIM_ * 2);
    unsigned short* wqb  = (unsigned short*)take((size_t)KEYD_ * DIM_ * 2);
    unsigned short* wkb  = (unsigned short*)take((size_t)KEYD_ * DIM_ * 2);
    unsigned short* wvb  = (unsigned short*)take((size_t)VALD_ * DIM_ * 2);
    unsigned short* wrab = (unsigned short*)take((size_t)128 * DIM_ * 2);
    float* ra    = (float*)take((size_t)BT_ * 128 * 4);
    float* qb    = (float*)take((size_t)BT_ * KEYD_ * 4);
    float* kb    = (float*)take((size_t)BT_ * KEYD_ * 4);
    float* vb    = (float*)take((size_t)BT_ * VALD_ * 4);
    float* decay = (float*)take((size_t)BT_ * H_ * 4);
    int*   topi  = (int*)  take((size_t)BT_ * K_ * 4);
    float* rwb   = (float*)take((size_t)BT_ * K_ * 4);
    float* yslot = (float*)take((size_t)BT_ * K_ * HV_ * 4);

    const int CT = 256;
    f32_to_bf16_kernel<<<(BT_ * DIM_ + CT - 1) / CT, CT, 0, stream>>>(x, xb, BT_ * DIM_);
    f32_to_bf16_kernel<<<(KEYD_ * DIM_ + CT - 1) / CT, CT, 0, stream>>>(Wq, wqb, KEYD_ * DIM_);
    f32_to_bf16_kernel<<<(KEYD_ * DIM_ + CT - 1) / CT, CT, 0, stream>>>(Wk, wkb, KEYD_ * DIM_);
    f32_to_bf16_kernel<<<(VALD_ * DIM_ + CT - 1) / CT, CT, 0, stream>>>(Wv, wvb, VALD_ * DIM_);
    f32_to_bf16_kernel<<<(H_ * DIM_ + CT - 1) / CT, CT, 0, stream>>>(Wr, wrab, H_ * DIM_);
    f32_to_bf16_kernel<<<(H_ * DIM_ + CT - 1) / CT, CT, 0, stream>>>(Wa, wrab + (size_t)H_ * DIM_, H_ * DIM_);

    // router/decay GEMM: [BT x 1024] x [1024 x 128]
    gemm_bf16_wmma<<<(BT_ / BM_) * (128 / BN_), 256, 0, stream>>>(xb, wrab, ra, BT_, 128, DIM_);
    // q/k/v projections: [BT x 1024] x [1024 x 2048]
    gemm_bf16_wmma<<<(BT_ / BM_) * (KEYD_ / BN_), 256, 0, stream>>>(xb, wqb, qb, BT_, KEYD_, DIM_);
    gemm_bf16_wmma<<<(BT_ / BM_) * (KEYD_ / BN_), 256, 0, stream>>>(xb, wkb, kb, BT_, KEYD_, DIM_);
    gemm_bf16_wmma<<<(BT_ / BM_) * (VALD_ / BN_), 256, 0, stream>>>(xb, wvb, vb, BT_, VALD_, DIM_);

    router_kernel<<<BT_, 32, 0, stream>>>(ra, A_log, dt_bias, topi, rwb, decay);

    silu_l2norm_kernel<<<(BT_ * H_) / 8, 256, 0, stream>>>(qb, BT_ * H_);
    silu_l2norm_kernel<<<(BT_ * H_) / 8, 256, 0, stream>>>(kb, BT_ * H_);
    silu_kernel<<<(BT_ * VALD_ + CT - 1) / CT, CT, 0, stream>>>(vb, BT_ * VALD_);

    scan_kernel<<<B_ * K_, 32, 0, stream>>>(qb, kb, vb, decay, topi, rwb, S0, yslot, out_S);

    out_proj_kernel<<<BT_, 256, 0, stream>>>(yslot, Wo, out_y);
}
